// FastConformerBlock_82480551953121
// MI455X (gfx1250) — compile-verified, matrix operand order checked by reading, not run on hardware
//
#include <hip/hip_runtime.h>

// ---------------------------------------------------------------------------
// FastConformer block for MI455X (gfx1250, wave32).
// All GEMMs: bf16 x bf16 -> f32 via v_wmma_f32_16x16x32_bf16.
// GEMM tiles staged with global_load_async_to_lds_b128 (ASYNCcnt) and
// double-buffered LDS so async copies overlap WMMA compute.
// Wave tile 64x64 (16 WMMA per k-step per wave).
// ---------------------------------------------------------------------------

#define CB 4
#define CT 2048
#define CD 512
#define CH 8
#define CDH 64
#define CFF 2048
#define CK 9
#define CM (CB * CT)        // 8192 rows
#define CEPS 1e-5f

typedef __bf16 bf16;
typedef __attribute__((ext_vector_type(16))) __bf16 v16bf;
typedef __attribute__((ext_vector_type(8)))  float  v8f;

__device__ __forceinline__ v8f v8f_zero() {
    v8f z;
#pragma unroll
    for (int i = 0; i < 8; ++i) z[i] = 0.0f;
    return z;
}

__device__ __forceinline__ v8f wmma_bf16(v16bf a, v16bf b, v8f c) {
    return __builtin_amdgcn_wmma_f32_16x16x32_bf16(
        false, a, false, b, (short)0, c, false, false);
}

// 16x32 bf16 A-style fragment from LDS (row-major, `stride` bf16 per row).
// Also used for B when the B tile is stored N-major.
__device__ __forceinline__ v16bf frag_ld(const bf16* __restrict__ p, int stride, int lane) {
    const bf16* rp = p + (size_t)(lane & 15) * stride;
    const int hh = (lane >> 4) << 3;
    v16bf a;
#pragma unroll
    for (int v = 0; v < 8; ++v) {
        const int kk = 2 * v + hh + ((v >= 4) ? 8 : 0);
        a[2 * v]     = rp[kk];
        a[2 * v + 1] = rp[kk + 1];
    }
    return a;
}

// sigmoid with v_rcp_f32 (avoid the IEEE division refinement chain)
__device__ __forceinline__ float sigmoidf_fast(float x) {
    return __builtin_amdgcn_rcpf(1.0f + __expf(-x));
}

// CDNA5 async copy: 16 bytes global -> LDS, tracked by ASYNCcnt.
__device__ __forceinline__ void async_cp_b128(unsigned lds_off, const void* gptr) {
    asm volatile("global_load_async_to_lds_b128 %0, %1, off"
                 :: "v"(lds_off), "v"(gptr)
                 : "memory");
}
__device__ __forceinline__ void wait_async0() {
    asm volatile("s_wait_asynccnt 0x0" ::: "memory");
}
__device__ __forceinline__ unsigned lds_off32(const void* p) {
    return (unsigned)(unsigned long long)p;   // low 32 bits = LDS offset
}

// ---------------------------------------------------------------------------
// Weight conversion: W (K x N, f32) -> Wt (N x K, bf16)
// ---------------------------------------------------------------------------
__global__ void wtrans_kernel(const float* __restrict__ W, bf16* __restrict__ Wt,
                              int Kd, int N) {
    int i = blockIdx.x * blockDim.x + threadIdx.x;
    if (i >= Kd * N) return;
    const int n = i / Kd, k = i % Kd;
    Wt[i] = (bf16)W[(size_t)k * N + n];
}

// ---------------------------------------------------------------------------
// LayerNorm over D=512; optional f32 and bf16 outputs.
// ---------------------------------------------------------------------------
__global__ __launch_bounds__(256) void ln_kernel(const float* __restrict__ X,
                                                 const float* __restrict__ g,
                                                 const float* __restrict__ b,
                                                 float* __restrict__ Yf,
                                                 bf16* __restrict__ Ybf) {
    const int row = blockIdx.x;
    const int tid = threadIdx.x;
    __shared__ float red[256];
    const size_t base = (size_t)row * CD;
    float x0 = X[base + tid];
    float x1 = X[base + tid + 256];
    red[tid] = x0 + x1;
    __syncthreads();
#pragma unroll
    for (int s = 128; s > 0; s >>= 1) {
        if (tid < s) red[tid] += red[tid + s];
        __syncthreads();
    }
    const float mean = red[0] * (1.0f / CD);
    __syncthreads();
    const float d0 = x0 - mean, d1 = x1 - mean;
    red[tid] = d0 * d0 + d1 * d1;
    __syncthreads();
#pragma unroll
    for (int s = 128; s > 0; s >>= 1) {
        if (tid < s) red[tid] += red[tid + s];
        __syncthreads();
    }
    const float rstd = rsqrtf(red[0] * (1.0f / CD) + CEPS);
    const float y0 = d0 * rstd * g[tid] + b[tid];
    const float y1 = d1 * rstd * g[tid + 256] + b[tid + 256];
    if (Yf)  { Yf[base + tid] = y0;        Yf[base + tid + 256] = y1; }
    if (Ybf) { Ybf[base + tid] = (bf16)y0; Ybf[base + tid + 256] = (bf16)y1; }
}

// ---------------------------------------------------------------------------
// WMMA GEMM: C[M,N] = epi( A[M,K](bf16) @ Wt[N,K](bf16)^T + bias )
// Block tile 128x128, BK=32, 4 wave32 (2x2), wave tile 64x64.
// Double-buffered LDS, async global->LDS staging (8x b128 per thread/tile).
// act: 0 none, 1 SiLU. res: optional f32 residual (C = res + scale*result).
// Cf / Cbf: optional f32 / bf16 outputs.
// ---------------------------------------------------------------------------
#define BM 128
#define BN 128
#define BK 32

__global__ __launch_bounds__(128) void gemm_kernel(const bf16* __restrict__ A,
                                                   const bf16* __restrict__ Wt,
                                                   const float* __restrict__ bias,
                                                   const float* __restrict__ res,
                                                   float scale,
                                                   float* __restrict__ Cf,
                                                   bf16* __restrict__ Cbf,
                                                   int N, int Kd, int act) {
    __shared__ bf16 sA[2][BM * BK];   // 2 x 8 KB
    __shared__ bf16 sB[2][BN * BK];   // 2 x 8 KB (N-major)

    const int tid  = threadIdx.x;     // 0..127
    const int lane = tid & 31;
    const int wave = tid >> 5;        // 0..3
    const int wm   = wave >> 1;       // 0..1 -> 64-row strip
    const int wn   = wave & 1;        // 0..1 -> 64-col strip
    const int bm   = blockIdx.x * BM;
    const int bn   = blockIdx.y * BN;

    // staging: each thread copies one full 64B tile row per matrix
    const bf16* Asrc = A  + (size_t)(bm + tid) * Kd;
    const bf16* Bsrc = Wt + (size_t)(bn + tid) * Kd;
    unsigned aoff[2], boff[2];
    aoff[0] = lds_off32(&sA[0][tid * BK]);
    aoff[1] = lds_off32(&sA[1][tid * BK]);
    boff[0] = lds_off32(&sB[0][tid * BK]);
    boff[1] = lds_off32(&sB[1][tid * BK]);

    v8f acc[4][4];
#pragma unroll
    for (int mt = 0; mt < 4; ++mt)
#pragma unroll
        for (int nt = 0; nt < 4; ++nt) acc[mt][nt] = v8f_zero();

    // preload tile 0 (4x b128 per matrix per thread)
#pragma unroll
    for (int j = 0; j < 4; ++j) {
        async_cp_b128(aoff[0] + 16 * j, Asrc + 8 * j);
        async_cp_b128(boff[0] + 16 * j, Bsrc + 8 * j);
    }
    wait_async0();
    __syncthreads();

    int buf = 0;
    for (int k0 = 0; k0 < Kd; k0 += BK) {
        const bool more = (k0 + BK) < Kd;
        if (more) {
            const int nb = buf ^ 1;
            const bf16* an  = Asrc + k0 + BK;
            const bf16* bn2 = Bsrc + k0 + BK;
#pragma unroll
            for (int j = 0; j < 4; ++j) {
                async_cp_b128(aoff[nb] + 16 * j, an + 8 * j);
                async_cp_b128(boff[nb] + 16 * j, bn2 + 8 * j);
            }
            __builtin_prefetch(Asrc + k0 + 2 * BK, 0, 1);
            __builtin_prefetch(Bsrc + k0 + 2 * BK, 0, 1);
        }

        const bf16* cA = sA[buf];
        const bf16* cB = sB[buf];
        v16bf bfr[4];
#pragma unroll
        for (int nt = 0; nt < 4; ++nt)
            bfr[nt] = frag_ld(cB + (wn * 64 + nt * 16) * BK, BK, lane);
#pragma unroll
        for (int mt = 0; mt < 4; ++mt) {
            const v16bf af = frag_ld(cA + (wm * 64 + mt * 16) * BK, BK, lane);
#pragma unroll
            for (int nt = 0; nt < 4; ++nt)
                acc[mt][nt] = wmma_bf16(af, bfr[nt], acc[mt][nt]);
        }

        if (more) wait_async0();
        __syncthreads();
        buf ^= 1;
    }

    const int lm = lane & 15, lh = lane >> 4;
#pragma unroll
    for (int mt = 0; mt < 4; ++mt) {
#pragma unroll
        for (int nt = 0; nt < 4; ++nt) {
#pragma unroll
            for (int r = 0; r < 8; ++r) {
                const int row = bm + wm * 64 + mt * 16 + r + 8 * lh;
                const int col = bn + wn * 64 + nt * 16 + lm;
                float v = acc[mt][nt][r] + bias[col];
                if (act == 1) v = v * sigmoidf_fast(v);
                const size_t o = (size_t)row * N + col;
                if (res) v = res[o] + scale * v;
                if (Cf)  Cf[o] = v;
                if (Cbf) Cbf[o] = (bf16)v;
            }
        }
    }
}

// ---------------------------------------------------------------------------
// Flash attention (bf16 in/out): block = (b, h, 64 q-rows), 4 wave32s, dh=64.
// ---------------------------------------------------------------------------
__global__ __launch_bounds__(128) void attn_kernel(const bf16* __restrict__ Q,
                                                   const bf16* __restrict__ Kb,
                                                   const bf16* __restrict__ Vb,
                                                   bf16* __restrict__ O) {
    __shared__ bf16 sQ[64 * 64];        // (qrow, dh)
    __shared__ bf16 sK[64 * 64];        // (s, dh)
    __shared__ bf16 sV[64 * 64];        // (dh, s)  transposed
    __shared__ bf16 sP[4][16 * 64];     // per-wave P tile

    const int tid  = threadIdx.x;
    const int lane = tid & 31;
    const int wave = tid >> 5;
    const int q0   = blockIdx.x * 64;
    const int b    = blockIdx.y / CH;
    const int h    = blockIdx.y % CH;
    const size_t base = ((size_t)b * CT) * CD + (size_t)h * CDH;

#pragma unroll
    for (int i = 0; i < 4; ++i) {
        const int idx = tid + 128 * i;          // 512 uint4 slots
        const int r = idx >> 3, cv = idx & 7;
        ((uint4*)sQ)[idx] = *(const uint4*)(Q + base + (size_t)(q0 + r) * CD + cv * 8);
    }
    __syncthreads();
    const v16bf qf0 = frag_ld(sQ + wave * 16 * 64,      64, lane);
    const v16bf qf1 = frag_ld(sQ + wave * 16 * 64 + 32, 64, lane);

    v8f oacc[4];
#pragma unroll
    for (int nt = 0; nt < 4; ++nt) oacc[nt] = v8f_zero();
    float mrun[8], lrun[8];
#pragma unroll
    for (int r = 0; r < 8; ++r) { mrun[r] = -3.0e38f; lrun[r] = 0.0f; }

    const int lm = lane & 15, lh = lane >> 4;

    for (int s0 = 0; s0 < CT; s0 += 64) {
        // batch K/V tile loads into registers, then LDS stores
        uint4 kreg[4], vreg[4];
#pragma unroll
        for (int i = 0; i < 4; ++i) {
            const int idx = tid + 128 * i;
            const int r = idx >> 3, cv = idx & 7;
            const size_t ro = base + (size_t)(s0 + r) * CD + cv * 8;
            kreg[i] = *(const uint4*)(Kb + ro);
            vreg[i] = *(const uint4*)(Vb + ro);
        }
#pragma unroll
        for (int i = 0; i < 4; ++i) {
            const int idx = tid + 128 * i;
            const int r = idx >> 3, cv = idx & 7;
            ((uint4*)sK)[idx] = kreg[i];
            const bf16* e = reinterpret_cast<const bf16*>(&vreg[i]);
#pragma unroll
            for (int j = 0; j < 8; ++j) sV[(cv * 8 + j) * 64 + r] = e[j];
        }
        __syncthreads();

        v8f sfr[4];
#pragma unroll
        for (int nt = 0; nt < 4; ++nt) {
            const v16bf kb0 = frag_ld(sK + nt * 16 * 64,      64, lane);
            const v16bf kb1 = frag_ld(sK + nt * 16 * 64 + 32, 64, lane);
            v8f c = v8f_zero();
            c = wmma_bf16(qf0, kb0, c);
            c = wmma_bf16(qf1, kb1, c);
            sfr[nt] = c;
        }

#pragma unroll
        for (int r = 0; r < 8; ++r) {
            float mx = -3.0e38f;
#pragma unroll
            for (int nt = 0; nt < 4; ++nt) mx = fmaxf(mx, sfr[nt][r] * 0.125f);
#pragma unroll
            for (int m = 1; m < 16; m <<= 1) mx = fmaxf(mx, __shfl_xor(mx, m, 32));
            const float mnew = fmaxf(mrun[r], mx);
            const float corr = __expf(mrun[r] - mnew);
            float ssum = 0.0f;
#pragma unroll
            for (int nt = 0; nt < 4; ++nt) {
                const float p = __expf(sfr[nt][r] * 0.125f - mnew);
                sfr[nt][r] = p;
                ssum += p;
            }
#pragma unroll
            for (int m = 1; m < 16; m <<= 1) ssum += __shfl_xor(ssum, m, 32);
            lrun[r] = lrun[r] * corr + ssum;
            mrun[r] = mnew;
#pragma unroll
            for (int nt = 0; nt < 4; ++nt) {
                oacc[nt][r] *= corr;
                sP[wave][(r + 8 * lh) * 64 + nt * 16 + lm] = (bf16)sfr[nt][r];
            }
        }
        __syncthreads();

        const v16bf pf0 = frag_ld(sP[wave],      64, lane);
        const v16bf pf1 = frag_ld(sP[wave] + 32, 64, lane);
#pragma unroll
        for (int nt = 0; nt < 4; ++nt) {
            const v16bf vb0 = frag_ld(sV + nt * 16 * 64,      64, lane);
            const v16bf vb1 = frag_ld(sV + nt * 16 * 64 + 32, 64, lane);
            oacc[nt] = wmma_bf16(pf0, vb0, oacc[nt]);
            oacc[nt] = wmma_bf16(pf1, vb1, oacc[nt]);
        }
        __syncthreads();
    }

#pragma unroll
    for (int r = 0; r < 8; ++r) {
        const float inv = __builtin_amdgcn_rcpf(lrun[r]);
        const int row = q0 + wave * 16 + r + 8 * lh;
#pragma unroll
        for (int nt = 0; nt < 4; ++nt)
            O[base + (size_t)row * CD + nt * 16 + lm] = (bf16)(oacc[nt][r] * inv);
    }
}

// ---------------------------------------------------------------------------
// Elementwise kernels
// ---------------------------------------------------------------------------
__global__ void glu_kernel(const float* __restrict__ in, float* __restrict__ out) {
    int i = blockIdx.x * blockDim.x + threadIdx.x;       // over M*D
    if (i >= CM * CD) return;
    const int row = i / CD, d = i % CD;
    const float a = in[(size_t)row * (2 * CD) + d];
    const float g = in[(size_t)row * (2 * CD) + CD + d];
    out[i] = a * sigmoidf_fast(g);
}

__global__ void dwconv_kernel(const float* __restrict__ in,
                              const float* __restrict__ w,
                              const float* __restrict__ b,
                              float* __restrict__ out) {
    int i = blockIdx.x * blockDim.x + threadIdx.x;       // over B*T*D
    if (i >= CB * CT * CD) return;
    const int d = i % CD;
    const int t = (i / CD) % CT;
    const int bb = i / (CD * CT);
    float acc = b[d];
#pragma unroll
    for (int k = 0; k < CK; ++k) {
        const int tt = t + k - (CK - 1) / 2;
        if (tt >= 0 && tt < CT)
            acc += in[((size_t)bb * CT + tt) * CD + d] * w[d * CK + k];
    }
    out[i] = acc;
}

__global__ __launch_bounds__(256) void bnstats_kernel(const float* __restrict__ X,
                                                      float* __restrict__ stats) {
    const int d = blockIdx.x, tid = threadIdx.x;
    __shared__ float r1[256], r2[256];
    float s = 0.0f, s2 = 0.0f;
    for (int r = tid; r < CM; r += 256) {
        const float v = X[(size_t)r * CD + d];
        s += v; s2 += v * v;
    }
    r1[tid] = s; r2[tid] = s2;
    __syncthreads();
#pragma unroll
    for (int st = 128; st > 0; st >>= 1) {
        if (tid < st) { r1[tid] += r1[tid + st]; r2[tid] += r2[tid + st]; }
        __syncthreads();
    }
    if (tid == 0) {
        const float m = r1[0] * (1.0f / CM);
        stats[d] = m;
        stats[CD + d] = r2[0] * (1.0f / CM) - m * m;
    }
}

__global__ void bnsilu_kernel(const float* __restrict__ in,
                              const float* __restrict__ stats,
                              const float* __restrict__ g,
                              const float* __restrict__ b,
                              bf16* __restrict__ out) {
    int i = blockIdx.x * blockDim.x + threadIdx.x;       // over M*D
    if (i >= CM * CD) return;
    const int d = i % CD;
    const float v = (in[i] - stats[d]) * rsqrtf(stats[CD + d] + CEPS) * g[d] + b[d];
    out[i] = (bf16)(v * sigmoidf_fast(v));
}

// ---------------------------------------------------------------------------
// Host pipeline
// ---------------------------------------------------------------------------
extern "C" void kernel_launch(void* const* d_in, const int* in_sizes, int n_in,
                              void* d_out, int out_size, void* d_ws, size_t ws_size,
                              hipStream_t stream) {
    (void)in_sizes; (void)n_in; (void)out_size; (void)ws_size;
    const float* x        = (const float*)d_in[0];
    const float* ff1_ln_g = (const float*)d_in[1];
    const float* ff1_ln_b = (const float*)d_in[2];
    const float* ff1_w1   = (const float*)d_in[3];
    const float* ff1_b1   = (const float*)d_in[4];
    const float* ff1_w2   = (const float*)d_in[5];
    const float* ff1_b2   = (const float*)d_in[6];
    const float* wq = (const float*)d_in[7];  const float* bq = (const float*)d_in[8];
    const float* wk = (const float*)d_in[9];  const float* bk = (const float*)d_in[10];
    const float* wv = (const float*)d_in[11]; const float* bv = (const float*)d_in[12];
    const float* wo = (const float*)d_in[13]; const float* bo = (const float*)d_in[14];
    const float* conv_ln_g = (const float*)d_in[15];
    const float* conv_ln_b = (const float*)d_in[16];
    const float* pw1_w = (const float*)d_in[17]; const float* pw1_b = (const float*)d_in[18];
    const float* dw_w  = (const float*)d_in[19]; const float* dw_b  = (const float*)d_in[20];
    const float* bn_g  = (const float*)d_in[21]; const float* bn_b  = (const float*)d_in[22];
    const float* pw2_w = (const float*)d_in[23]; const float* pw2_b = (const float*)d_in[24];
    const float* ff2_ln_g = (const float*)d_in[25];
    const float* ff2_ln_b = (const float*)d_in[26];
    const float* ff2_w1   = (const float*)d_in[27];
    const float* ff2_b1   = (const float*)d_in[28];
    const float* ff2_w2   = (const float*)d_in[29];
    const float* ff2_b2   = (const float*)d_in[30];
    const float* fin_ln_g = (const float*)d_in[31];
    const float* fin_ln_b = (const float*)d_in[32];

    char* ws = (char*)d_ws;
    const size_t MB = 1024ull * 1024ull;
    float* X    = (float*)(ws + 0 * MB);      // 16 MB residual stream (f32)
    bf16*  Xbf  = (bf16*)(ws + 16 * MB);      // 8 MB bf16 copy of X for QKV
    bf16*  LNbf = (bf16*)(ws + 24 * MB);      // 8 MB LN output (bf16)
    bf16*  Hbf  = (bf16*)(ws + 32 * MB);      // 32 MB FFN hidden (bf16)
    float* Hf   = (float*)(ws + 64 * MB);     // 32 MB pw1 output (f32)
    bf16*  Qb   = (bf16*)(ws + 96 * MB);      // 8 MB
    bf16*  Kbuf = (bf16*)(ws + 104 * MB);     // 8 MB
    bf16*  Vbuf = (bf16*)(ws + 112 * MB);     // 8 MB
    bf16*  AObf = (bf16*)(ws + 120 * MB);     // 8 MB attention out (bf16)
    float* G1   = (float*)(ws + 128 * MB);    // 16 MB GLU out (f32)
    float* H2   = (float*)(ws + 144 * MB);    // 16 MB dwconv out (f32)
    bf16*  T2bf = (bf16*)(ws + 160 * MB);     // 8 MB bnsilu out (bf16)
    float* BNs  = (float*)(ws + 168 * MB);    // 4 KB batchnorm stats
    bf16*  wbf  = (bf16*)(ws + 168 * MB + 4096);

    bf16* t_ff1w1 = wbf;                       // (FF, D)
    bf16* t_ff1w2 = t_ff1w1 + CFF * CD;        // (D, FF)
    bf16* t_wq    = t_ff1w2 + CD * CFF;
    bf16* t_wk    = t_wq + CD * CD;
    bf16* t_wv    = t_wk + CD * CD;
    bf16* t_wo    = t_wv + CD * CD;
    bf16* t_pw1   = t_wo + CD * CD;            // (2D, D)
    bf16* t_pw2   = t_pw1 + 2 * CD * CD;       // (D, D)
    bf16* t_ff2w1 = t_pw2 + CD * CD;
    bf16* t_ff2w2 = t_ff2w1 + CFF * CD;

    auto wt = [&](const float* W, bf16* Wt, int Kd, int N) {
        const int n = Kd * N;
        wtrans_kernel<<<(n + 255) / 256, 256, 0, stream>>>(W, Wt, Kd, N);
    };
    wt(ff1_w1, t_ff1w1, CD, CFF);
    wt(ff1_w2, t_ff1w2, CFF, CD);
    wt(wq, t_wq, CD, CD); wt(wk, t_wk, CD, CD);
    wt(wv, t_wv, CD, CD); wt(wo, t_wo, CD, CD);
    wt(pw1_w, t_pw1, CD, 2 * CD);
    wt(pw2_w, t_pw2, CD, CD);
    wt(ff2_w1, t_ff2w1, CD, CFF);
    wt(ff2_w2, t_ff2w2, CFF, CD);

    auto gemm = [&](const bf16* A, const bf16* Wt, const float* bias,
                    const float* res, float scale, float* Cf, bf16* Cbf,
                    int N, int Kd, int act) {
        dim3 grid(CM / BM, N / BN);
        gemm_kernel<<<grid, 128, 0, stream>>>(A, Wt, bias, res, scale, Cf, Cbf, N, Kd, act);
    };

    const int nElem = CM * CD;
    const dim3 eGrid((nElem + 255) / 256);

    // ---- FF1: x = x + 0.5 * (silu(ln(x) @ w1 + b1) @ w2 + b2) ----
    ln_kernel<<<CM, 256, 0, stream>>>(x, ff1_ln_g, ff1_ln_b, nullptr, LNbf);
    gemm(LNbf, t_ff1w1, ff1_b1, nullptr, 0.0f, nullptr, Hbf, CFF, CD, 1);
    gemm(Hbf, t_ff1w2, ff1_b2, x, 0.5f, X, Xbf, CD, CFF, 0);

    // ---- Attention ----
    gemm(Xbf, t_wq, bq, nullptr, 0.0f, nullptr, Qb,   CD, CD, 0);
    gemm(Xbf, t_wk, bk, nullptr, 0.0f, nullptr, Kbuf, CD, CD, 0);
    gemm(Xbf, t_wv, bv, nullptr, 0.0f, nullptr, Vbuf, CD, CD, 0);
    attn_kernel<<<dim3(CT / 64, CB * CH), 128, 0, stream>>>(Qb, Kbuf, Vbuf, AObf);
    gemm(AObf, t_wo, bo, X, 1.0f, X, nullptr, CD, CD, 0);

    // ---- Conv module ----
    ln_kernel<<<CM, 256, 0, stream>>>(X, conv_ln_g, conv_ln_b, nullptr, LNbf);
    gemm(LNbf, t_pw1, pw1_b, nullptr, 0.0f, Hf, nullptr, 2 * CD, CD, 0);
    glu_kernel<<<eGrid, 256, 0, stream>>>(Hf, G1);
    dwconv_kernel<<<eGrid, 256, 0, stream>>>(G1, dw_w, dw_b, H2);
    bnstats_kernel<<<CD, 256, 0, stream>>>(H2, BNs);
    bnsilu_kernel<<<eGrid, 256, 0, stream>>>(H2, BNs, bn_g, bn_b, T2bf);
    gemm(T2bf, t_pw2, pw2_b, X, 1.0f, X, nullptr, CD, CD, 0);

    // ---- FF2 ----
    ln_kernel<<<CM, 256, 0, stream>>>(X, ff2_ln_g, ff2_ln_b, nullptr, LNbf);
    gemm(LNbf, t_ff2w1, ff2_b1, nullptr, 0.0f, nullptr, Hbf, CFF, CD, 1);
    gemm(Hbf, t_ff2w2, ff2_b2, X, 0.5f, X, nullptr, CD, CFF, 0);

    // ---- Final LN -> output ----
    ln_kernel<<<CM, 256, 0, stream>>>(X, fin_ln_g, fin_ln_b, (float*)d_out, nullptr);
}
